// AttentionGCN_28638841930195
// MI455X (gfx1250) — matrix operand order repaired and checked by reading, not compiled, not run on hardware
//
#include <hip/hip_runtime.h>

#define F_IN   8
#define PSTEPS 12
#define OUTF   32

typedef __attribute__((ext_vector_type(16))) _Float16 v16h;
typedef __attribute__((ext_vector_type(8)))  _Float16 v8h;
typedef __attribute__((ext_vector_type(2)))  __fp16   v2hp;  // cvt_pkrtz return type
typedef __attribute__((ext_vector_type(8)))  float    v8f;

union U16H {
  v16h v;
  v8h  q[2];
  v2hp p[8];
};

__device__ __forceinline__ void atomAdd(float* p, float v) {
  (void)__hip_atomic_fetch_add(p, v, __ATOMIC_RELAXED, __HIP_MEMORY_SCOPE_AGENT);
}

// Fast activations on the native transcendental units (v_exp_f32 / v_rcp_f32).
__device__ __forceinline__ float fast_sigmoid(float x) {
  return __builtin_amdgcn_rcpf(1.f + __expf(-x));
}
__device__ __forceinline__ float fast_tanh(float x) {
  float t = __expf(2.f * x);                    // overflow -> inf -> rcp -> 0 -> 1.0
  return 1.f - 2.f * __builtin_amdgcn_rcpf(t + 1.f);
}

__device__ __forceinline__ v8f wmma_f16(v16h a, v16h b, v8f c) {
  // D = A(16x32 f16) * B(32x16 f16) + C(16x16 f32)
  return __builtin_amdgcn_wmma_f32_16x16x32_f16(
      /*neg_a=*/false, a, /*neg_b=*/false, b,
      /*c_mod=*/(short)0, c, /*reuse_a=*/false, /*reuse_b=*/false);
}

// Build A operand (16x32 f16 tile) from row-major [N x 32] f32 matrix.
// ISA 16-bit A layout: lane = M row (lane%16); elements 0..7 -> K = hh*8+0..7,
// elements 8..15 -> K = 16+hh*8+0..7. Packed f32->f16 via v_cvt_pk_rtz_f16_f32.
__device__ __forceinline__ v16h loadA_g(const float* __restrict__ M,
                                        int nb, int j, int hh, int N) {
  int row = nb + j;
  if (row >= N) { v16h z = {}; return z; }
  const float4* r0 = (const float4*)(M + (size_t)row * OUTF + hh * 8);
  const float4* r1 = (const float4*)(M + (size_t)row * OUTF + 16 + hh * 8);
  float4 f0 = r0[0], f1 = r0[1], f2 = r1[0], f3 = r1[1];
  U16H u;
  u.p[0] = __builtin_amdgcn_cvt_pkrtz(f0.x, f0.y);
  u.p[1] = __builtin_amdgcn_cvt_pkrtz(f0.z, f0.w);
  u.p[2] = __builtin_amdgcn_cvt_pkrtz(f1.x, f1.y);
  u.p[3] = __builtin_amdgcn_cvt_pkrtz(f1.z, f1.w);
  u.p[4] = __builtin_amdgcn_cvt_pkrtz(f2.x, f2.y);
  u.p[5] = __builtin_amdgcn_cvt_pkrtz(f2.z, f2.w);
  u.p[6] = __builtin_amdgcn_cvt_pkrtz(f3.x, f3.y);
  u.p[7] = __builtin_amdgcn_cvt_pkrtz(f3.z, f3.w);
  return u.v;
}

// Same A layout from an LDS 16x32 f32 tile (rows always valid; OOB rows hold 0).
__device__ __forceinline__ v16h loadA_lds(const float* r32, int j, int hh) {
  const float4* r0 = (const float4*)(r32 + j * OUTF + hh * 8);
  const float4* r1 = (const float4*)(r32 + j * OUTF + 16 + hh * 8);
  float4 f0 = r0[0], f1 = r0[1], f2 = r1[0], f3 = r1[1];
  U16H u;
  u.p[0] = __builtin_amdgcn_cvt_pkrtz(f0.x, f0.y);
  u.p[1] = __builtin_amdgcn_cvt_pkrtz(f0.z, f0.w);
  u.p[2] = __builtin_amdgcn_cvt_pkrtz(f1.x, f1.y);
  u.p[3] = __builtin_amdgcn_cvt_pkrtz(f1.z, f1.w);
  u.p[4] = __builtin_amdgcn_cvt_pkrtz(f2.x, f2.y);
  u.p[5] = __builtin_amdgcn_cvt_pkrtz(f2.z, f2.w);
  u.p[6] = __builtin_amdgcn_cvt_pkrtz(f3.x, f3.y);
  u.p[7] = __builtin_amdgcn_cvt_pkrtz(f3.z, f3.w);
  return u.v;
}

// Packed-B LDS layout: row index = ((k0sel*2 + hh)*32 + col), 16 halfs per row,
// padded to 24 halfs (48B) so the 16 lanes of a half-wave hit distinct banks.
// Element e of the row = L[(k0sel*32 + (e/8)*16 + hh*8 + e%8)*32 + col] as f16.
#define BROW 24

__device__ __forceinline__ v16h loadB(const _Float16* sB, int k0sel, int hh, int col) {
  const _Float16* ptr = sB + (((k0sel * 2 + hh) * 32) + col) * BROW;
  U16H u;
  u.q[0] = *(const v8h*)(ptr);
  u.q[1] = *(const v8h*)(ptr + 8);
  return u.v;
}

// ---------------- setup kernels ----------------

__global__ void probs_kernel(const float* __restrict__ att, float* __restrict__ probs) {
  if (threadIdx.x == 0 && blockIdx.x == 0) {
    float m = -1e30f;
    for (int i = 0; i < PSTEPS; ++i) m = fmaxf(m, att[i]);
    float e[PSTEPS], s = 0.f;
    for (int i = 0; i < PSTEPS; ++i) { e[i] = expf(att[i] - m); s += e[i]; }
    for (int i = 0; i < PSTEPS; ++i) probs[i] = e[i] / s;
  }
}

__global__ void deg_init(float* __restrict__ deg, int N) {
  int i = blockIdx.x * blockDim.x + threadIdx.x;
  if (i < N) deg[i] = 1.0f;   // self-loop weight
}

__global__ void deg_scatter(const int* __restrict__ ei, const float* __restrict__ w,
                            float* __restrict__ deg, int E) {
  int e = blockIdx.x * blockDim.x + threadIdx.x;
  if (e < E) atomAdd(&deg[ei[E + e]], w[e]);
}

__global__ void deg_finalize(float* __restrict__ dinv, int N) {
  int i = blockIdx.x * blockDim.x + threadIdx.x;
  if (i < N) dinv[i] = rsqrtf(fmaxf(dinv[i], 1e-30f));  // deg >= 1 always
}

__global__ void norm_kernel(const int* __restrict__ ei, const float* __restrict__ w,
                            const float* __restrict__ dinv, float* __restrict__ nrm, int E) {
  int e = blockIdx.x * blockDim.x + threadIdx.x;
  if (e < E) nrm[e] = dinv[ei[e]] * w[e] * dinv[ei[E + e]];
}

__global__ void zero2(float* __restrict__ a, float* __restrict__ b, int n) {
  int i = blockIdx.x * blockDim.x + threadIdx.x;
  if (i < n) { a[i] = 0.f; b[i] = 0.f; }
}

// ---------------- per-timestep kernels ----------------

// Fused: XW = Xt @ W (stored for the edge gather) and Agg = bias + dinv^2 * XW
// (self-loop term; full overwrite so no pre-zero). One wave per node: n is
// wave-uniform (forced via readfirstlane) so x[n,f,t] / dinv[n] become scalar loads.
__global__ void xw_agg_kernel(const float* __restrict__ x,
                              const float* __restrict__ Wz, const float* __restrict__ Wr,
                              const float* __restrict__ Wh,
                              const float* __restrict__ bz, const float* __restrict__ br,
                              const float* __restrict__ bh, const float* __restrict__ dinv,
                              float* __restrict__ XWz, float* __restrict__ XWr,
                              float* __restrict__ XWh,
                              float* __restrict__ Az, float* __restrict__ Ar,
                              float* __restrict__ Ah, int t, int N) {
  int wave = threadIdx.x >> 5, o = threadIdx.x & 31;
  int n = blockIdx.x * 8 + wave;
  if (n >= N) return;                       // wave-uniform branch
  n = __builtin_amdgcn_readfirstlane(n);
  float xv[F_IN];
#pragma unroll
  for (int f = 0; f < F_IN; ++f) xv[f] = x[(n * F_IN + f) * PSTEPS + t];
  float az = 0.f, ar = 0.f, ah = 0.f;
#pragma unroll
  for (int f = 0; f < F_IN; ++f) {
    az += xv[f] * Wz[f * OUTF + o];
    ar += xv[f] * Wr[f * OUTF + o];
    ah += xv[f] * Wh[f * OUTF + o];
  }
  int i = n * OUTF + o;
  float s2 = dinv[n] * dinv[n];
  XWz[i] = az; XWr[i] = ar; XWh[i] = ah;
  Az[i] = bz[o] + s2 * az;
  Ar[i] = br[o] + s2 * ar;
  Ah[i] = bh[o] + s2 * ah;
}

// One wave32 per edge: lane = feature. Gather XW[src], scale by norm,
// atomic-scatter to Agg[dst] (all node matrices are L2-resident; the streamed
// ei/nrm arrays are the HBM traffic -> prefetch them ahead).
__global__ void scatter_kernel(const int* __restrict__ ei, const float* __restrict__ nrm,
                               const float* __restrict__ XWz, const float* __restrict__ XWr,
                               const float* __restrict__ XWh,
                               float* __restrict__ Az, float* __restrict__ Ar,
                               float* __restrict__ Ah, int E) {
  long long i = (long long)blockIdx.x * blockDim.x + threadIdx.x;
  int e = (int)(i >> 5), o = (int)(i & 31);
  if (e >= E) return;
  if (o == 0 && e + 4096 < E) {
    __builtin_prefetch(ei + e + 4096, 0, 1);       // global_prefetch_b8
    __builtin_prefetch(ei + E + e + 4096, 0, 1);
    __builtin_prefetch(nrm + e + 4096, 0, 1);
  }
  int s = ei[e], d = ei[E + e];
  float nm = nrm[e];
  atomAdd(&Az[d * OUTF + o], nm * XWz[s * OUTF + o]);
  atomAdd(&Ar[d * OUTF + o], nm * XWr[s * OUTF + o]);
  atomAdd(&Ah[d * OUTF + o], nm * XWh[s * OUTF + o]);
}

// GRU cell via WMMA: each wave owns a 16-node tile; K=64 (agg | H) split as 2 x32 chunks.
__global__ void __launch_bounds__(256) gru_kernel(
    const float* __restrict__ Az, const float* __restrict__ Ar, const float* __restrict__ Ah,
    float* __restrict__ H, float* __restrict__ Hacc,
    const float* __restrict__ Lz, const float* __restrict__ Lr, const float* __restrict__ Lh,
    const float* __restrict__ lbz, const float* __restrict__ lbr, const float* __restrict__ lbh,
    const float* __restrict__ probs, int t, int N) {
  __shared__ __align__(16) _Float16 sBz[128 * BROW];  // packed-B weights, 6KB each
  __shared__ __align__(16) _Float16 sBr[128 * BROW];
  __shared__ __align__(16) _Float16 sBh[128 * BROW];
  __shared__ __align__(16) float    sRH[8][16 * OUTF]; // per-wave R*H transpose staging

  int tid = threadIdx.x;
  // Pack L (64x32 f32, row-major, K-major rows) into the B register layout.
  for (int i = tid; i < 2048; i += 256) {
    int e = i & 15;
    int row = i >> 4;          // ((k0sel*2 + hh)*32 + c)
    int c = row & 31;
    int hhk = row >> 5;        // k0sel*2 + hh
    int K = (hhk >> 1) * 32 + (e >> 3) * 16 + (hhk & 1) * 8 + (e & 7);
    sBz[row * BROW + e] = (_Float16)Lz[K * OUTF + c];
    sBr[row * BROW + e] = (_Float16)Lr[K * OUTF + c];
    sBh[row * BROW + e] = (_Float16)Lh[K * OUTF + c];
  }
  __syncthreads();

  int wave = tid >> 5, lane = tid & 31;
  int hh = lane >> 4, j = lane & 15;
  int nb = blockIdx.x * 128 + wave * 16;
  float p = probs[t];

  // Shared A chunks for Z and R gates
  v16h aH = loadA_g(H, nb, j, hh, N);
  v16h aZ = loadA_g(Az, nb, j, hh, N);
  v16h aR = loadA_g(Ar, nb, j, hh, N);

  v8f zc0 = {}, zc1 = {}, rc0 = {}, rc1 = {};
  zc0 = wmma_f16(aZ, loadB(sBz, 0, hh, j),      zc0);
  zc0 = wmma_f16(aH, loadB(sBz, 1, hh, j),      zc0);
  zc1 = wmma_f16(aZ, loadB(sBz, 0, hh, j + 16), zc1);
  zc1 = wmma_f16(aH, loadB(sBz, 1, hh, j + 16), zc1);
  rc0 = wmma_f16(aR, loadB(sBr, 0, hh, j),      rc0);
  rc0 = wmma_f16(aH, loadB(sBr, 1, hh, j),      rc0);
  rc1 = wmma_f16(aR, loadB(sBr, 0, hh, j + 16), rc1);
  rc1 = wmma_f16(aH, loadB(sBr, 1, hh, j + 16), rc1);

  float bz0 = lbz[j], bz1 = lbz[j + 16];
  float br0 = lbr[j], br1 = lbr[j + 16];

  // C/D layout: vgpr v, lane (hh,j) holds row M = v + 8*hh, col N = j (+16 tile 1)
  float h0[8], h1[8], z0[8], z1[8];
#pragma unroll
  for (int v = 0; v < 8; ++v) {
    int row = v + 8 * hh;
    int node = nb + row;
    float hv0 = 0.f, hv1 = 0.f;
    if (node < N) { hv0 = H[node * OUTF + j]; hv1 = H[node * OUTF + 16 + j]; }
    h0[v] = hv0; h1[v] = hv1;
    z0[v] = fast_sigmoid(zc0[v] + bz0);
    z1[v] = fast_sigmoid(zc1[v] + bz1);
    float r0 = fast_sigmoid(rc0[v] + br0);
    float r1 = fast_sigmoid(rc1[v] + br1);
    sRH[wave][row * OUTF + j]      = r0 * hv0;   // OOB rows naturally 0 (hv==0)
    sRH[wave][row * OUTF + 16 + j] = r1 * hv1;
  }
  __syncthreads();

  v16h aHt1 = loadA_g(Ah, nb, j, hh, N);
  v16h aHt2 = loadA_lds(&sRH[wave][0], j, hh);
  v8f hc0 = {}, hc1 = {};
  hc0 = wmma_f16(aHt1, loadB(sBh, 0, hh, j),      hc0);
  hc0 = wmma_f16(aHt2, loadB(sBh, 1, hh, j),      hc0);
  hc1 = wmma_f16(aHt1, loadB(sBh, 0, hh, j + 16), hc1);
  hc1 = wmma_f16(aHt2, loadB(sBh, 1, hh, j + 16), hc1);

  float bh0 = lbh[j], bh1 = lbh[j + 16];
#pragma unroll
  for (int v = 0; v < 8; ++v) {
    int node = nb + v + 8 * hh;
    if (node < N) {
      float ht0 = fast_tanh(hc0[v] + bh0);
      float ht1 = fast_tanh(hc1[v] + bh1);
      float hn0 = z0[v] * h0[v] + (1.f - z0[v]) * ht0;
      float hn1 = z1[v] * h1[v] + (1.f - z1[v]) * ht1;
      H[node * OUTF + j]      = hn0;
      H[node * OUTF + 16 + j] = hn1;
      Hacc[node * OUTF + j]      += p * hn0;
      Hacc[node * OUTF + 16 + j] += p * hn1;
    }
  }
}

// out[n] = relu( relu(Hacc[n]) . Wp + bp + x[n,1,P-1] )
__global__ void out_kernel(const float* __restrict__ Hacc, const float* __restrict__ Wp,
                           const float* __restrict__ bp, const float* __restrict__ x,
                           float* __restrict__ out, int N) {
  int n = blockIdx.x * blockDim.x + threadIdx.x;
  if (n >= N) return;
  float acc = bp[0];
#pragma unroll
  for (int o = 0; o < OUTF; ++o) acc += fmaxf(Hacc[n * OUTF + o], 0.f) * Wp[o];
  acc += x[(n * F_IN + 1) * PSTEPS + (PSTEPS - 1)];
  out[n] = fmaxf(acc, 0.f);
}

// ---------------- launcher ----------------

extern "C" void kernel_launch(void* const* d_in, const int* in_sizes, int n_in,
                              void* d_out, int out_size, void* d_ws, size_t ws_size,
                              hipStream_t stream) {
  (void)n_in; (void)out_size; (void)ws_size;
  const float* x   = (const float*)d_in[0];
  const int*   ei  = (const int*)d_in[1];    // (2,E) int32: row0 src, row1 dst
  const float* w   = (const float*)d_in[2];
  const float* Wz  = (const float*)d_in[3];
  const float* bz  = (const float*)d_in[4];
  const float* Wr  = (const float*)d_in[5];
  const float* br  = (const float*)d_in[6];
  const float* Wh  = (const float*)d_in[7];
  const float* bh  = (const float*)d_in[8];
  const float* Lz  = (const float*)d_in[9];
  const float* lbz = (const float*)d_in[10];
  const float* Lr  = (const float*)d_in[11];
  const float* lbr = (const float*)d_in[12];
  const float* Lh  = (const float*)d_in[13];
  const float* lbh = (const float*)d_in[14];
  const float* att = (const float*)d_in[15];
  const float* Wp  = (const float*)d_in[16];
  const float* bp  = (const float*)d_in[17];
  float* out = (float*)d_out;

  const int N = in_sizes[0] / (F_IN * PSTEPS);
  const int E = in_sizes[2];

  float* ws   = (float*)d_ws;
  float* dinv = ws;  ws += N;
  float* nrm  = ws;  ws += E;
  float* XWz  = ws;  ws += (size_t)N * OUTF;
  float* XWr  = ws;  ws += (size_t)N * OUTF;
  float* XWh  = ws;  ws += (size_t)N * OUTF;
  float* Az   = ws;  ws += (size_t)N * OUTF;
  float* Ar   = ws;  ws += (size_t)N * OUTF;
  float* Ah   = ws;  ws += (size_t)N * OUTF;
  float* H    = ws;  ws += (size_t)N * OUTF;
  float* Hacc = ws;  ws += (size_t)N * OUTF;
  float* probs = ws; ws += 16;

  const int T = 256;
  auto cdiv = [](long long a, long long b) { return (unsigned)((a + b - 1) / b); };

  probs_kernel<<<1, 32, 0, stream>>>(att, probs);
  deg_init<<<cdiv(N, T), T, 0, stream>>>(dinv, N);
  deg_scatter<<<cdiv(E, T), T, 0, stream>>>(ei, w, dinv, E);
  deg_finalize<<<cdiv(N, T), T, 0, stream>>>(dinv, N);
  norm_kernel<<<cdiv(E, T), T, 0, stream>>>(ei, w, dinv, nrm, E);
  zero2<<<cdiv((long long)N * OUTF, T), T, 0, stream>>>(H, Hacc, N * OUTF);

  for (int t = 0; t < PSTEPS; ++t) {
    xw_agg_kernel<<<cdiv(N, 8), T, 0, stream>>>(x, Wz, Wr, Wh, bz, br, bh, dinv,
                                                XWz, XWr, XWh, Az, Ar, Ah, t, N);
    scatter_kernel<<<cdiv((long long)E * 32, T), T, 0, stream>>>(ei, nrm, XWz, XWr, XWh,
                                                                 Az, Ar, Ah, E);
    gru_kernel<<<cdiv(N, 128), T, 0, stream>>>(Az, Ar, Ah, H, Hacc, Lz, Lr, Lh,
                                               lbz, lbr, lbh, probs, t, N);
  }
  out_kernel<<<cdiv(N, T), T, 0, stream>>>(Hacc, Wp, bp, x, out, N);
}
